// GdnBlock_45105746542918
// MI455X (gfx1250) — compile-verified, matrix operand order checked by reading, not run on hardware
//
#include <hip/hip_runtime.h>
#include <hip/hip_bf16.h>

// ---------------------------------------------------------------------------
// GatedDeltaNet block for MI455X (gfx1250, wave32, WMMA + TDM async staging)
// B=4 T=1024 HIDDEN=1024 H=16 DK=64 DV=128 KEY=1024 VAL=2048 CONV=4
// ---------------------------------------------------------------------------

typedef __attribute__((ext_vector_type(16))) __bf16 bf16x16;
typedef __attribute__((ext_vector_type(8)))  __bf16 bf16x8;
typedef __attribute__((ext_vector_type(8)))  float  f32x8;
typedef __attribute__((ext_vector_type(4)))  unsigned int u32x4;
typedef __attribute__((ext_vector_type(8)))  int i32x8;
typedef __attribute__((ext_vector_type(4)))  int i32x4;

// TDM builtin arity differs between toolchains; the 6-arg form ships with the
// therock-10.0 headers (amdgpu-toolchain), the 5-arg form with ROCm 7.2.
#if defined(__has_include)
#if __has_include(<hip/amd_detail/amd_gfx1250_TDM.h>)
#define GDN_TDM_6ARG 1
#endif
#endif
#ifndef GDN_TDM_6ARG
#define GDN_TDM_6ARG 0
#endif

#define B_    4
#define T_    1024
#define HID_  1024
#define NH_   16
#define DK_   64
#define DV_   128
#define KEY_  1024
#define VAL_  2048
#define CONV_ 4
#define CACHE_ 147456              // H*DK*DV + CONV*(KEY+KEY+VAL)
#define COFF_Q 131072              // H*DK*DV
#define COFF_K 135168
#define COFF_V 139264

#define BROW_ 40                   // LDS row stride in bf16 (64B data + 16B pad)

union bfrag { bf16x16 v; bf16x8 h[2]; };

// --------------------------- fp32 -> bf16 convert --------------------------
__global__ void cvt_bf16_kernel(const float* __restrict__ in,
                                __bf16* __restrict__ out, int n) {
  int i = blockIdx.x * blockDim.x + threadIdx.x;
  if (i < n) out[i] = (__bf16)in[i];
}

// transpose + convert: in[K,N] fp32 row-major -> out[N,K] bf16
__global__ void transpose_bf16_kernel(const float* __restrict__ in,
                                      __bf16* __restrict__ out, int K, int N) {
  int i = blockIdx.x * blockDim.x + threadIdx.x;
  if (i >= K * N) return;
  int n = i % N;
  int k = i / N;
  out[(size_t)n * K + k] = (__bf16)in[i];
}

// ------------------- bf16 WMMA GEMM with TDM B-staging ---------------------
// C[M,N] fp32 = A[M,K] bf16 x BT[N,K]^T bf16.
// Block = 256 threads = 8 waves (2 along M x 4 along N); wave computes
// 16(M) x 64(N) via 4 accumulators. M%32==0, N%256==0, K%32==0.
// The 256x32 bf16 B tile is DMA'd into LDS by the Tensor Data Mover
// (double buffered, one descriptor per k-step, wave 0 issues/waits),
// with D# padding (16B per 64B row) to spread LDS banks. A is register
// double-buffered from global. All four B fragments are pulled from LDS
// into distinct registers before the WMMA group so the four WMMAs issue
// back-to-back behind a single s_wait_dscnt. EXEC is all-ones at every WMMA.
__global__ __launch_bounds__(256)
void gemm_bf16_tdm_kernel(const __bf16* __restrict__ A,
                          const __bf16* __restrict__ BT,
                          float* __restrict__ C, int M, int N, int K) {
  __shared__ __bf16 Bsh[2][256 * BROW_];   // 2 x 20KB

  const int wave = threadIdx.x >> 5;
  const int lane = threadIdx.x & 31;
  const int m0 = blockIdx.x * 32 + (wave >> 2) * 16;
  const int nblk = blockIdx.y * 256;
  const int nloc = (wave & 3) * 64;

  // 16-bit A/B fragment layout (ISA 7.12.2): lane&15 selects row/col,
  // lane>=16 shifts the K window by 8; elems 0..7 = K kb..kb+7,
  // elems 8..15 = K kb+16..kb+23.
  const int rc = lane & 15;
  const int kb = (lane >> 4) << 3;

  const __bf16* ap = A + (size_t)(m0 + rc) * K + kb;

  // Issue one TDM descriptor: load 256(N) x 32(K) bf16 tile of BT into LDS.
  auto issue_tdm = [&](int buf, int k0) {
    unsigned long long ga =
        (unsigned long long)(const void*)(BT + (size_t)nblk * K + k0);
    unsigned lds = (unsigned)(unsigned long long)&Bsh[buf][0];
    u32x4 g0;
    g0[0] = 1u;                                    // count=1 (valid D#)
    g0[1] = lds;                                   // lds_addr (bytes)
    g0[2] = (unsigned)(ga & 0xFFFFFFFFu);          // global_addr[31:0]
    g0[3] = (unsigned)((ga >> 32) & 0x1FFFFFFu)    // global_addr[56:32]
            | (2u << 30);                          // type=2 ("image")
    i32x8 g1;
    g1[0] = (int)((1u << 16)                       // data_size = 2 bytes
                  | (1u << 20)                     // pad_enable
                  | (3u << 22)                     // pad_interval: 16 DWORDs
                  | (3u << 25));                   // pad_amount: 4 DWORDs
    g1[1] = (int)(((unsigned)K & 0xFFFFu) << 16);  // tensor_dim0 lo16
    g1[2] = (int)(((unsigned)K >> 16) | (256u << 16)); // td0 hi | td1 lo
    g1[3] = (int)(32u << 16);                      // td1 hi | tile_dim0=32
    g1[4] = 256;                                   // tile_dim1=256, tile_dim2=0
    g1[5] = K;                                     // tensor_dim0_stride lo32
    g1[6] = 0;                                     // stride0 hi | stride1 lo
    g1[7] = 0;                                     // stride1 hi
    i32x4 gz = {0, 0, 0, 0};
#if GDN_TDM_6ARG
    i32x8 gz8 = {0, 0, 0, 0, 0, 0, 0, 0};
    __builtin_amdgcn_tensor_load_to_lds(g0, g1, gz, gz, gz8, 0);
#else
    __builtin_amdgcn_tensor_load_to_lds(g0, g1, gz, gz, 0);
#endif
  };

  // prologue: stage first B tile, prefetch first A fragment
  if (wave == 0) issue_tdm(0, 0);
  bfrag a0;
  a0.h[0] = *(const bf16x8*)(ap);
  a0.h[1] = *(const bf16x8*)(ap + 16);
  if (wave == 0) __builtin_amdgcn_s_wait_tensorcnt(0);
  __syncthreads();

  f32x8 acc[4] = {};
  int buf = 0;
  for (int k0 = 0; k0 < K; k0 += 32) {
    const bool more = (k0 + 32 < K);
    if (wave == 0 && more) issue_tdm(buf ^ 1, k0 + 32);

    // prefetch next A fragment (dummy re-read of k=0 on last iter)
    const int kn = more ? (k0 + 32) : 0;
    bfrag a1;
    a1.h[0] = *(const bf16x8*)(ap + kn);
    a1.h[1] = *(const bf16x8*)(ap + kn + 16);

    // pull all 4 B fragments from LDS first (distinct registers), then
    // issue the 4 WMMAs back-to-back behind one dscnt wait
    const __bf16* bsh = &Bsh[buf][0];
    bfrag bf[4];
#pragma unroll
    for (int tn = 0; tn < 4; ++tn) {
      const __bf16* brow = bsh + (size_t)(nloc + tn * 16 + rc) * BROW_ + kb;
      bf[tn].h[0] = *(const bf16x8*)(brow);
      bf[tn].h[1] = *(const bf16x8*)(brow + 16);
    }
#pragma unroll
    for (int tn = 0; tn < 4; ++tn)
      acc[tn] = __builtin_amdgcn_wmma_f32_16x16x32_bf16(
          false, a0.v, false, bf[tn].v, (short)0, acc[tn], false, false);
    a0 = a1;

    if (wave == 0) __builtin_amdgcn_s_wait_tensorcnt(0);
    __syncthreads();
    buf ^= 1;
  }

  // 32-bit C/D layout: elem i -> M = i + (lane>=16 ? 8 : 0), N = lane&15
  const int rsel = (lane >> 4) << 3;
#pragma unroll
  for (int tn = 0; tn < 4; ++tn)
#pragma unroll
    for (int i = 0; i < 8; ++i)
      C[(size_t)(m0 + rsel + i) * N + (nblk + nloc + tn * 16 + rc)] = acc[tn][i];
}

// --------------------------- bf16 WMMA GEMM (N == 16) ----------------------
// 8 waves laid out along M; each wave one 16x16 tile. M%128==0, K%32==0.
__global__ __launch_bounds__(256)
void gemm_bf16_n16_kernel(const __bf16* __restrict__ A,
                          const __bf16* __restrict__ BT,
                          float* __restrict__ C, int M, int K) {
  const int wave = threadIdx.x >> 5;
  const int lane = threadIdx.x & 31;
  const int m0 = blockIdx.x * 128 + wave * 16;
  const int rc = lane & 15;
  const int kb = (lane >> 4) << 3;

  const __bf16* ap = A + (size_t)(m0 + rc) * K + kb;
  const __bf16* bp = BT + (size_t)rc * K + kb;

  bfrag a0, b0;
  a0.h[0] = *(const bf16x8*)(ap);
  a0.h[1] = *(const bf16x8*)(ap + 16);
  b0.h[0] = *(const bf16x8*)(bp);
  b0.h[1] = *(const bf16x8*)(bp + 16);

  f32x8 acc = {};
  for (int k0 = 0; k0 < K; k0 += 32) {
    const int kn = (k0 + 32 < K) ? (k0 + 32) : 0;
    bfrag a1, b1;
    a1.h[0] = *(const bf16x8*)(ap + kn);
    a1.h[1] = *(const bf16x8*)(ap + kn + 16);
    b1.h[0] = *(const bf16x8*)(bp + kn);
    b1.h[1] = *(const bf16x8*)(bp + kn + 16);
    acc = __builtin_amdgcn_wmma_f32_16x16x32_bf16(
        false, a0.v, false, b0.v, (short)0, acc, false, false);
    a0 = a1;
    b0 = b1;
  }

  const int rsel = (lane >> 4) << 3;
#pragma unroll
  for (int i = 0; i < 8; ++i)
    C[(size_t)(m0 + rsel + i) * 16 + rc] = acc[i];
}

// --------------------- depthwise causal conv4 + SiLU -----------------------
__global__ void conv_silu_kernel(const float* __restrict__ pre,
                                 const float* __restrict__ w,
                                 const float* __restrict__ cache_in, int coff,
                                 int ch, float* __restrict__ out) {
  int idx = blockIdx.x * blockDim.x + threadIdx.x;
  if (idx >= B_ * T_ * ch) return;
  int c = idx % ch;
  int t = (idx / ch) % T_;
  int b = idx / (ch * T_);
  float acc = 0.f;
#pragma unroll
  for (int i = 0; i < CONV_; ++i) {
    int j = t + i - (CONV_ - 1);
    float xv = (j >= 0)
                   ? pre[((size_t)(b * T_ + j)) * ch + c]
                   : cache_in[(size_t)b * CACHE_ + coff + c * CONV_ + (1 + t + i)];
    acc += xv * w[c * CONV_ + i];
  }
  out[idx] = acc / (1.f + __expf(-acc));   // SiLU
}

// new conv cache = last CONV pre-conv inputs, written to output cache region
__global__ void conv_state_kernel(const float* __restrict__ pre,
                                  float* __restrict__ cache_out, int coff,
                                  int ch) {
  int idx = blockIdx.x * blockDim.x + threadIdx.x;
  if (idx >= B_ * ch * CONV_) return;
  int j = idx % CONV_;
  int c = (idx / CONV_) % ch;
  int b = idx / (CONV_ * ch);
  cache_out[(size_t)b * CACHE_ + coff + c * CONV_ + j] =
      pre[((size_t)(b * T_ + (T_ - CONV_ + j))) * ch + c];
}

// -------------------- per-head l2 normalize (DK=64) ------------------------
// x viewed as [B*T*H, 64]; one wave per head, 2 elements per lane
__global__ __launch_bounds__(256)
void l2norm_kernel(float* __restrict__ x, float scale) {
  int wave = threadIdx.x >> 5;
  int lane = threadIdx.x & 31;
  size_t head = (size_t)blockIdx.x * 8 + wave;
  float2 v = *(const float2*)(x + head * 64 + lane * 2);
  float s = v.x * v.x + v.y * v.y;
#pragma unroll
  for (int o = 16; o > 0; o >>= 1) s += __shfl_xor(s, o, 32);
  float r = rsqrtf(s + 1e-6f) * scale;
  v.x *= r;
  v.y *= r;
  *(float2*)(x + head * 64 + lane * 2) = v;
}

// ------------------------ beta / log-decay g -------------------------------
__global__ void beta_g_kernel(const float* __restrict__ bpre,
                              const float* __restrict__ apre,
                              const float* __restrict__ A_log,
                              const float* __restrict__ dt_bias,
                              float* __restrict__ beta, float* __restrict__ g,
                              int n) {
  int i = blockIdx.x * blockDim.x + threadIdx.x;
  if (i >= n) return;
  int h = i % NH_;
  beta[i] = 1.f / (1.f + __expf(-bpre[i]));
  float xv = apre[i] + dt_bias[h];
  float sp = (xv > 20.f) ? xv : __logf(1.f + __expf(xv));
  g[i] = -__expf(A_log[h]) * sp;
}

// ------------------- gated delta-rule recurrent scan -----------------------
// grid = B*H blocks, 128 threads; thread owns one DV column, S[64] in VGPRs.
// q/k/beta/g addresses are block-uniform -> backend scalarizes to s_load.
__global__ __launch_bounds__(128)
void scan_kernel(const float* __restrict__ q, const float* __restrict__ k,
                 const float* __restrict__ v, const float* __restrict__ beta,
                 const float* __restrict__ g, const float* __restrict__ cache_in,
                 float* __restrict__ o, float* __restrict__ cache_out) {
  const int b = blockIdx.x >> 4;
  const int h = blockIdx.x & 15;
  const int dv = threadIdx.x;

  float S[DK_];
  const float* S0 = cache_in + (size_t)b * CACHE_ + (size_t)h * DK_ * DV_;
#pragma unroll
  for (int dk = 0; dk < DK_; ++dk) S[dk] = S0[dk * DV_ + dv];

  for (int t = 0; t < T_; ++t) {
    const size_t bt = (size_t)(b * T_ + t);
    const float* kp = k + bt * KEY_ + h * DK_;
    const float* qp = q + bt * KEY_ + h * DK_;
    const float bv = beta[bt * NH_ + h];
    const float eg = __expf(g[bt * NH_ + h]);
    const float vt = v[bt * VAL_ + h * DV_ + dv];

    float kS = 0.f;
#pragma unroll
    for (int dk = 0; dk < DK_; ++dk) kS += kp[dk] * S[dk];
    const float verr = (vt - kS * eg) * bv;

    float ot = 0.f;
#pragma unroll
    for (int dk = 0; dk < DK_; ++dk) {
      float sv = S[dk] * eg + kp[dk] * verr;
      S[dk] = sv;
      ot += qp[dk] * sv;
    }
    o[bt * VAL_ + h * DV_ + dv] = ot;
  }

  float* Sout = cache_out + (size_t)b * CACHE_ + (size_t)h * DK_ * DV_;
#pragma unroll
  for (int dk = 0; dk < DK_; ++dk) Sout[dk * DV_ + dv] = S[dk];
}

// ----------------- gated RMSNorm + SiLU gate -> bf16 o ---------------------
// o viewed as [B*T*H, 128]; one wave per head, 4 elements per lane
__global__ __launch_bounds__(256)
void norm_gate_kernel(const float* __restrict__ o, const float* __restrict__ gpre,
                      const float* __restrict__ norm_w,
                      __bf16* __restrict__ obf) {
  int wave = threadIdx.x >> 5;
  int lane = threadIdx.x & 31;
  size_t head = (size_t)blockIdx.x * 8 + wave;
  size_t base = head * DV_ + lane * 4;
  float4 ov = *(const float4*)(o + base);
  float s = ov.x * ov.x + ov.y * ov.y + ov.z * ov.z + ov.w * ov.w;
#pragma unroll
  for (int off = 16; off > 0; off >>= 1) s += __shfl_xor(s, off, 32);
  float r = rsqrtf(s * (1.f / (float)DV_) + 1e-6f);
  float4 gv = *(const float4*)(gpre + base);
  int dv = lane * 4;
  float n0 = norm_w[dv], n1 = norm_w[dv + 1], n2 = norm_w[dv + 2], n3 = norm_w[dv + 3];
  auto silu = [](float x) { return x / (1.f + __expf(-x)); };
  obf[base + 0] = (__bf16)(ov.x * r * n0 * silu(gv.x));
  obf[base + 1] = (__bf16)(ov.y * r * n1 * silu(gv.y));
  obf[base + 2] = (__bf16)(ov.z * r * n2 * silu(gv.z));
  obf[base + 3] = (__bf16)(ov.w * r * n3 * silu(gv.w));
}

// ---------------------------------------------------------------------------
extern "C" void kernel_launch(void* const* d_in, const int* in_sizes, int n_in,
                              void* d_out, int out_size, void* d_ws, size_t ws_size,
                              hipStream_t stream) {
  const float* x        = (const float*)d_in[0];
  // d_in[1] = attn_mask (unused)
  const float* rnn_in   = (const float*)d_in[2];
  const float* Wq       = (const float*)d_in[3];
  const float* Wk       = (const float*)d_in[4];
  const float* Wv       = (const float*)d_in[5];
  const float* Wb       = (const float*)d_in[6];
  const float* Wa       = (const float*)d_in[7];
  const float* Wg       = (const float*)d_in[8];
  const float* Wo       = (const float*)d_in[9];
  const float* conv_q_w = (const float*)d_in[10];
  const float* conv_k_w = (const float*)d_in[11];
  const float* conv_v_w = (const float*)d_in[12];
  const float* A_log    = (const float*)d_in[13];
  const float* dt_bias  = (const float*)d_in[14];
  const float* norm_w   = (const float*)d_in[15];

  float* out_main  = (float*)d_out;                           // [B,T,HID]
  float* cache_out = (float*)d_out + (size_t)B_ * T_ * HID_;  // [B,CACHE]

  // ---- carve workspace (256B aligned slabs) ----
  char* wp = (char*)d_ws;
  auto alloc = [&](size_t bytes) {
    void* p = (void*)wp;
    wp += (bytes + 255) & ~(size_t)255;
    return p;
  };
  const size_t BT = (size_t)B_ * T_;
  __bf16* xbf  = (__bf16*)alloc(BT * HID_ * 2);
  __bf16* WqT  = (__bf16*)alloc((size_t)KEY_ * HID_ * 2);
  __bf16* WkT  = (__bf16*)alloc((size_t)KEY_ * HID_ * 2);
  __bf16* WvT  = (__bf16*)alloc((size_t)VAL_ * HID_ * 2);
  __bf16* WgT  = (__bf16*)alloc((size_t)VAL_ * HID_ * 2);
  __bf16* WbT  = (__bf16*)alloc((size_t)NH_ * HID_ * 2);
  __bf16* WaT  = (__bf16*)alloc((size_t)NH_ * HID_ * 2);
  __bf16* WoT  = (__bf16*)alloc((size_t)HID_ * VAL_ * 2);
  float* qpre  = (float*)alloc(BT * KEY_ * 4);   // reused as o[0:16M)
  float* kpre  = (float*)alloc(BT * KEY_ * 4);   // reused as o[16M:32M)
  float* vpre  = (float*)alloc(BT * VAL_ * 4);   // reused as obf
  float* bpre  = (float*)alloc(BT * NH_ * 4);
  float* apre  = (float*)alloc(BT * NH_ * 4);
  float* gpre  = (float*)alloc(BT * VAL_ * 4);
  float* qbuf  = (float*)alloc(BT * KEY_ * 4);
  float* kbuf  = (float*)alloc(BT * KEY_ * 4);
  float* vbuf  = (float*)alloc(BT * VAL_ * 4);
  float* betab = (float*)alloc(BT * NH_ * 4);
  float* gbuf  = (float*)alloc(BT * NH_ * 4);
  float*  obuf = qpre;              // [B,T,H,DV] fp32  (qpre+kpre dead by then)
  __bf16* obf  = (__bf16*)vpre;     // [B,T,H,DV] bf16  (vpre dead by then)
  (void)ws_size; (void)n_in; (void)in_sizes; (void)out_size;

  auto cdiv = [](size_t a, size_t b) { return (unsigned)((a + b - 1) / b); };

  // 1) convert activations + transposed bf16 weights
  cvt_bf16_kernel<<<cdiv(BT * HID_, 256), 256, 0, stream>>>(x, xbf, (int)(BT * HID_));
  transpose_bf16_kernel<<<cdiv((size_t)HID_ * KEY_, 256), 256, 0, stream>>>(Wq, WqT, HID_, KEY_);
  transpose_bf16_kernel<<<cdiv((size_t)HID_ * KEY_, 256), 256, 0, stream>>>(Wk, WkT, HID_, KEY_);
  transpose_bf16_kernel<<<cdiv((size_t)HID_ * VAL_, 256), 256, 0, stream>>>(Wv, WvT, HID_, VAL_);
  transpose_bf16_kernel<<<cdiv((size_t)HID_ * VAL_, 256), 256, 0, stream>>>(Wg, WgT, HID_, VAL_);
  transpose_bf16_kernel<<<cdiv((size_t)HID_ * NH_, 256), 256, 0, stream>>>(Wb, WbT, HID_, NH_);
  transpose_bf16_kernel<<<cdiv((size_t)HID_ * NH_, 256), 256, 0, stream>>>(Wa, WaT, HID_, NH_);
  transpose_bf16_kernel<<<cdiv((size_t)VAL_ * HID_, 256), 256, 0, stream>>>(Wo, WoT, VAL_, HID_);

  // 2) projection GEMMs (WMMA bf16 -> fp32), TDM-staged B tiles
  const int M = (int)BT;                     // 4096
  dim3 blk(256);
  gemm_bf16_tdm_kernel<<<dim3(M / 32, KEY_ / 256), blk, 0, stream>>>(xbf, WqT, qpre, M, KEY_, HID_);
  gemm_bf16_tdm_kernel<<<dim3(M / 32, KEY_ / 256), blk, 0, stream>>>(xbf, WkT, kpre, M, KEY_, HID_);
  gemm_bf16_tdm_kernel<<<dim3(M / 32, VAL_ / 256), blk, 0, stream>>>(xbf, WvT, vpre, M, VAL_, HID_);
  gemm_bf16_tdm_kernel<<<dim3(M / 32, VAL_ / 256), blk, 0, stream>>>(xbf, WgT, gpre, M, VAL_, HID_);
  gemm_bf16_n16_kernel<<<dim3(M / 128), blk, 0, stream>>>(xbf, WbT, bpre, M, HID_);
  gemm_bf16_n16_kernel<<<dim3(M / 128), blk, 0, stream>>>(xbf, WaT, apre, M, HID_);

  // 3) causal conv + SiLU, and new conv caches
  conv_silu_kernel<<<cdiv(BT * KEY_, 256), 256, 0, stream>>>(qpre, conv_q_w, rnn_in, COFF_Q, KEY_, qbuf);
  conv_silu_kernel<<<cdiv(BT * KEY_, 256), 256, 0, stream>>>(kpre, conv_k_w, rnn_in, COFF_K, KEY_, kbuf);
  conv_silu_kernel<<<cdiv(BT * VAL_, 256), 256, 0, stream>>>(vpre, conv_v_w, rnn_in, COFF_V, VAL_, vbuf);
  conv_state_kernel<<<cdiv((size_t)B_ * KEY_ * CONV_, 256), 256, 0, stream>>>(qpre, cache_out, COFF_Q, KEY_);
  conv_state_kernel<<<cdiv((size_t)B_ * KEY_ * CONV_, 256), 256, 0, stream>>>(kpre, cache_out, COFF_K, KEY_);
  conv_state_kernel<<<cdiv((size_t)B_ * VAL_ * CONV_, 256), 256, 0, stream>>>(vpre, cache_out, COFF_V, VAL_);

  // 4) per-head l2norm (q also scaled by DK^-0.5), beta/g
  l2norm_kernel<<<cdiv(BT * NH_, 8), 256, 0, stream>>>(qbuf, 0.125f);
  l2norm_kernel<<<cdiv(BT * NH_, 8), 256, 0, stream>>>(kbuf, 1.0f);
  beta_g_kernel<<<cdiv(BT * NH_, 256), 256, 0, stream>>>(bpre, apre, A_log, dt_bias,
                                                         betab, gbuf, (int)(BT * NH_));

  // 5) gated delta-rule scan (writes o and S_final)
  scan_kernel<<<dim3(B_ * NH_), dim3(128), 0, stream>>>(qbuf, kbuf, vbuf, betab, gbuf,
                                                        rnn_in, obuf, cache_out);

  // 6) gated RMSNorm -> bf16, then output projection straight into d_out
  norm_gate_kernel<<<cdiv(BT * NH_, 8), 256, 0, stream>>>(obuf, gpre, norm_w, obf);
  gemm_bf16_tdm_kernel<<<dim3(M / 32, HID_ / 256), blk, 0, stream>>>(obf, WoT, out_main, M, HID_, VAL_);
}
